// ParticleInteractionBlock_55173149884911
// MI455X (gfx1250) — matrix operand order repaired and checked by reading, compiled, not run on hardware
//
#include <hip/hip_runtime.h>

#define NNODES 50000
#define NEDGES 600000
#define HD 128

typedef __attribute__((ext_vector_type(16))) __bf16 v16bf;
typedef __attribute__((ext_vector_type(8)))  __bf16 v8bf;
typedef __attribute__((ext_vector_type(8)))  float  v8f;
typedef __attribute__((ext_vector_type(4)))  float  f32x4;
typedef __attribute__((ext_vector_type(4)))  unsigned int u32x4;

union Frag {
  v16bf v;
  v8bf  h[2];
  u32x4 q[2];
};

__device__ __forceinline__ v8f vzero() {
  v8f z = {0.f,0.f,0.f,0.f,0.f,0.f,0.f,0.f};
  return z;
}

__device__ __forceinline__ v8f wmma_bf16(v16bf a, v16bf b, v8f c) {
  // D = A(16x32 bf16) * B(32x16 bf16) + C(16x16 f32)
  return __builtin_amdgcn_wmma_f32_16x16x32_bf16(false, a, false, b, (short)0, c, false, false);
}

// Load 8 consecutive f32 and convert to 8 bf16 (one half of an A fragment).
__device__ __forceinline__ v8bf cvt8(const float* __restrict__ p) {
  f32x4 a = *(const f32x4*)p;
  f32x4 b = *(const f32x4*)(p + 4);
  v8bf r;
  r[0] = (__bf16)a[0]; r[1] = (__bf16)a[1]; r[2] = (__bf16)a[2]; r[3] = (__bf16)a[3];
  r[4] = (__bf16)b[0]; r[5] = (__bf16)b[1]; r[6] = (__bf16)b[2]; r[7] = (__bf16)b[3];
  return r;
}

// Cooperative copy of one transposed bf16 weight matrix [128][K] from global
// into LDS with padded row stride K+8 (bank-conflict-avoiding).
template<int K>
__device__ __forceinline__ void stage_w(unsigned short* dst,
                                        const unsigned short* __restrict__ src,
                                        int tid) {
  constexpr int CH = K / 8;     // 16B chunks per row
  constexpr int LD = K + 8;     // padded stride (elements)
  for (int idx = tid; idx < 128 * CH; idx += 256) {
    int r = idx / CH;
    int c = idx - r * CH;
    *(u32x4*)(dst + r * LD + c * 8) = *(const u32x4*)(src + (size_t)r * K + c * 8);
  }
}

// Batched B-fragment load: all 8 N-tiles for one k-step.
template<int LDW>
__device__ __forceinline__ void load_b8(Frag b[8], const unsigned short* __restrict__ wlds,
                                        int k0, int l16, int hl) {
#pragma unroll
  for (int nt = 0; nt < 8; ++nt) {
    const unsigned short* p = wlds + (nt * 16 + l16) * LDW + k0 + hl * 8;
    b[nt].q[0] = *(const u32x4*)p;
    b[nt].q[1] = *(const u32x4*)(p + 16);
  }
}

// One 16x128 stripe times 128x128 weights, K=128, A from LDS (stride 136),
// B from LDS weight buffer (stride 136). B double-buffered across k-steps so
// the WMMA burst of step s overlaps the ds_loads for step s+1.
__device__ __forceinline__ void gemm_k128(const unsigned short* __restrict__ arow,
                                          const unsigned short* __restrict__ wlds,
                                          v8f acc[8], int l16, int hl) {
  const int koff = hl * 8;
  Frag bb[2][8];
  load_b8<136>(bb[0], wlds, 0, l16, hl);
#pragma unroll
  for (int s = 0; s < 4; ++s) {
    const int k0 = s * 32;
    if (s + 1 < 4) load_b8<136>(bb[(s + 1) & 1], wlds, k0 + 32, l16, hl);
    Frag a;
    a.q[0] = *(const u32x4*)(arow + k0 + koff);
    a.q[1] = *(const u32x4*)(arow + k0 + 16 + koff);
#pragma unroll
    for (int nt = 0; nt < 8; ++nt) acc[nt] = wmma_bf16(a.v, bb[s & 1][nt].v, acc[nt]);
  }
}

// bias + ReLU, write bf16 activations into LDS activation buffer (stride 136).
__device__ __forceinline__ void bias_relu_store(v8f acc[8], const float* __restrict__ bias,
                                                unsigned short* albs, int wv, int l16, int hl) {
#pragma unroll
  for (int nt = 0; nt < 8; ++nt) {
    float bb = bias[nt * 16 + l16];
#pragma unroll
    for (int i = 0; i < 8; ++i) {
      float v = acc[nt][i] + bb;
      v = fmaxf(v, 0.0f);
      albs[(wv * 16 + i + 8 * hl) * 136 + nt * 16 + l16] =
          __builtin_bit_cast(unsigned short, (__bf16)v);
    }
  }
}

// fp32 [K][128] -> bf16 transposed [128][K]
__global__ __launch_bounds__(256) void wcvt_kernel(const float* __restrict__ w,
                                                   unsigned short* __restrict__ wt, int K) {
  int idx = blockIdx.x * 256 + threadIdx.x;
  if (idx >= K * 128) return;
  int n = idx / K;
  int k = idx - n * K;
  float v = w[(size_t)k * 128 + n];
  wt[idx] = __builtin_bit_cast(unsigned short, (__bf16)v);
}

// ---------------- Edge MLP: [E,384] -> 128 -> 128 -> 128 + LN, scatter-add ----------------
__global__ __launch_bounds__(256, 2) void edge_mlp_kernel(
    const float* __restrict__ x, const float* __restrict__ e,
    const long long* __restrict__ eidx,
    const unsigned short* __restrict__ we0t, const unsigned short* __restrict__ we1t,
    const unsigned short* __restrict__ we2t,
    const float* __restrict__ be0, const float* __restrict__ be1, const float* __restrict__ be2,
    const float* __restrict__ ge, const float* __restrict__ bge,
    float* __restrict__ enew, float* agg) {
  extern __shared__ __align__(16) unsigned short smem[];
  unsigned short* wlds = smem;              // up to 128*392 bf16
  unsigned short* albs = smem + 128 * 392;  // 128*136 bf16

  const int tid  = threadIdx.x;
  const int lane = tid & 31;
  const int wv   = tid >> 5;       // wave id 0..7 -> 16-row stripe
  const int hl   = lane >> 4;      // lane half (K split)
  const int l16  = lane & 15;
  const int koff = hl * 8;
  const int base = blockIdx.x * 128;

  const long long* srcI = eidx;            // edge_index[0] = src
  const long long* dstI = eidx + NEDGES;   // edge_index[1] = dst

  const int row_e = base + wv * 16 + l16;
  const int re = row_e < NEDGES ? row_e : NEDGES - 1;
  const long long dr = dstI[re];
  const long long sr = srcI[re];
  const float* pd = x + (size_t)dr * HD;
  const float* ps = x + (size_t)sr * HD;
  const float* pe = e + (size_t)re * HD;

  stage_w<384>(wlds, we0t, tid);
  __syncthreads();

  v8f acc[8];
#pragma unroll
  for (int nt = 0; nt < 8; ++nt) acc[nt] = vzero();

  // Layer 0: K=384 ([x[dst] | x[src] | e]), A gathered straight from global + cvt;
  // B fragments double-buffered across the 12 k-steps.
  {
    Frag bb[2][8];
    load_b8<392>(bb[0], wlds, 0, l16, hl);
#pragma unroll
    for (int s = 0; s < 12; ++s) {
      const int k0 = s * 32;
      if (s + 1 < 12) load_b8<392>(bb[(s + 1) & 1], wlds, k0 + 32, l16, hl);
      const float* ap = (k0 < 128) ? (pd + k0) : (k0 < 256 ? (ps + k0 - 128) : (pe + k0 - 256));
      Frag a;
      a.h[0] = cvt8(ap + koff);
      a.h[1] = cvt8(ap + 16 + koff);
#pragma unroll
      for (int nt = 0; nt < 8; ++nt) acc[nt] = wmma_bf16(a.v, bb[s & 1][nt].v, acc[nt]);
    }
  }
  bias_relu_store(acc, be0, albs, wv, l16, hl);
  __syncthreads();
  stage_w<128>(wlds, we1t, tid);
  __syncthreads();

  const unsigned short* arow = albs + (wv * 16 + l16) * 136;

  // Layer 1
#pragma unroll
  for (int nt = 0; nt < 8; ++nt) acc[nt] = vzero();
  gemm_k128(arow, wlds, acc, l16, hl);
  bias_relu_store(acc, be1, albs, wv, l16, hl);
  __syncthreads();
  stage_w<128>(wlds, we2t, tid);
  __syncthreads();

  // Layer 2
#pragma unroll
  for (int nt = 0; nt < 8; ++nt) acc[nt] = vzero();
  gemm_k128(arow, wlds, acc, l16, hl);

  // bias + LayerNorm + stores + scatter-add
#pragma unroll
  for (int nt = 0; nt < 8; ++nt) {
    float bb = be2[nt * 16 + l16];
#pragma unroll
    for (int i = 0; i < 8; ++i) acc[nt][i] += bb;
  }
  float gg[8], b2[8];
#pragma unroll
  for (int nt = 0; nt < 8; ++nt) { gg[nt] = ge[nt * 16 + l16]; b2[nt] = bge[nt * 16 + l16]; }

#pragma unroll
  for (int i = 0; i < 8; ++i) {
    float s1 = 0.f, s2 = 0.f;
#pragma unroll
    for (int nt = 0; nt < 8; ++nt) { float v = acc[nt][i]; s1 += v; s2 += v * v; }
#pragma unroll
    for (int m = 1; m < 16; m <<= 1) { s1 += __shfl_xor(s1, m); s2 += __shfl_xor(s2, m); }
    float mu  = s1 * (1.0f / HD);
    float var = s2 * (1.0f / HD) - mu * mu;
    float rs  = rsqrtf(var + 1e-5f);
    int edge = base + wv * 16 + i + 8 * hl;
    bool ok  = edge < NEDGES;
    int ce   = ok ? edge : 0;
    long long d = dstI[ce];
#pragma unroll
    for (int nt = 0; nt < 8; ++nt) {
      int col = nt * 16 + l16;
      float o = (acc[nt][i] - mu) * rs * gg[nt] + b2[nt];
      if (ok) {
        enew[(size_t)edge * HD + col] = o;
        __hip_atomic_fetch_add(&agg[(size_t)d * HD + col], o,
                               __ATOMIC_RELAXED, __HIP_MEMORY_SCOPE_AGENT);
      }
    }
  }
}

// ---------------- Node MLP: [N,256] -> 128 -> 128 -> 128 + LN, residual ----------------
__global__ __launch_bounds__(256, 2) void node_mlp_kernel(
    const float* __restrict__ x,
    const unsigned short* __restrict__ wn0t, const unsigned short* __restrict__ wn1t,
    const unsigned short* __restrict__ wn2t,
    const float* __restrict__ bn0, const float* __restrict__ bn1, const float* __restrict__ bn2,
    const float* __restrict__ gn, const float* __restrict__ bgn,
    float* xa /* in: agg, out: x_new (same buffer) */) {
  extern __shared__ __align__(16) unsigned short smem[];
  unsigned short* wlds = smem;              // up to 128*264 bf16
  unsigned short* albs = smem + 128 * 264;  // 128*136 bf16

  const int tid  = threadIdx.x;
  const int lane = tid & 31;
  const int wv   = tid >> 5;
  const int hl   = lane >> 4;
  const int l16  = lane & 15;
  const int koff = hl * 8;
  const int base = blockIdx.x * 128;

  const int row_n = base + wv * 16 + l16;
  const int rn = row_n < NNODES ? row_n : NNODES - 1;
  const float* px = x  + (size_t)rn * HD;
  const float* pa = xa + (size_t)rn * HD;

  stage_w<256>(wlds, wn0t, tid);
  __syncthreads();

  v8f acc[8];
#pragma unroll
  for (int nt = 0; nt < 8; ++nt) acc[nt] = vzero();

  // Layer 0: K=256 ([x | agg]); B double-buffered across the 8 k-steps.
  {
    Frag bb[2][8];
    load_b8<264>(bb[0], wlds, 0, l16, hl);
#pragma unroll
    for (int s = 0; s < 8; ++s) {
      const int k0 = s * 32;
      if (s + 1 < 8) load_b8<264>(bb[(s + 1) & 1], wlds, k0 + 32, l16, hl);
      const float* ap = (k0 < 128) ? (px + k0) : (pa + k0 - 128);
      Frag a;
      a.h[0] = cvt8(ap + koff);
      a.h[1] = cvt8(ap + 16 + koff);
#pragma unroll
      for (int nt = 0; nt < 8; ++nt) acc[nt] = wmma_bf16(a.v, bb[s & 1][nt].v, acc[nt]);
    }
  }
  bias_relu_store(acc, bn0, albs, wv, l16, hl);
  __syncthreads();
  stage_w<128>(wlds, wn1t, tid);
  __syncthreads();

  const unsigned short* arow = albs + (wv * 16 + l16) * 136;

#pragma unroll
  for (int nt = 0; nt < 8; ++nt) acc[nt] = vzero();
  gemm_k128(arow, wlds, acc, l16, hl);
  bias_relu_store(acc, bn1, albs, wv, l16, hl);
  __syncthreads();
  stage_w<128>(wlds, wn2t, tid);
  __syncthreads();

#pragma unroll
  for (int nt = 0; nt < 8; ++nt) acc[nt] = vzero();
  gemm_k128(arow, wlds, acc, l16, hl);

#pragma unroll
  for (int nt = 0; nt < 8; ++nt) {
    float bb = bn2[nt * 16 + l16];
#pragma unroll
    for (int i = 0; i < 8; ++i) acc[nt][i] += bb;
  }
  float gg[8], b2[8];
#pragma unroll
  for (int nt = 0; nt < 8; ++nt) { gg[nt] = gn[nt * 16 + l16]; b2[nt] = bgn[nt * 16 + l16]; }

#pragma unroll
  for (int i = 0; i < 8; ++i) {
    float s1 = 0.f, s2 = 0.f;
#pragma unroll
    for (int nt = 0; nt < 8; ++nt) { float v = acc[nt][i]; s1 += v; s2 += v * v; }
#pragma unroll
    for (int m = 1; m < 16; m <<= 1) { s1 += __shfl_xor(s1, m); s2 += __shfl_xor(s2, m); }
    float mu  = s1 * (1.0f / HD);
    float var = s2 * (1.0f / HD) - mu * mu;
    float rs  = rsqrtf(var + 1e-5f);
    int node = base + wv * 16 + i + 8 * hl;
    bool ok  = node < NNODES;
#pragma unroll
    for (int nt = 0; nt < 8; ++nt) {
      int col = nt * 16 + l16;
      float o = (acc[nt][i] - mu) * rs * gg[nt] + b2[nt];
      if (ok) {
        size_t off = (size_t)node * HD + col;
        xa[off] = x[off] + o;   // residual; overwrites agg in place
      }
    }
  }
}

extern "C" void kernel_launch(void* const* d_in, const int* in_sizes, int n_in,
                              void* d_out, int out_size, void* d_ws, size_t ws_size,
                              hipStream_t stream) {
  (void)in_sizes; (void)n_in; (void)out_size; (void)ws_size;
  const float* x   = (const float*)d_in[0];
  const float* e   = (const float*)d_in[1];
  const long long* eidx = (const long long*)d_in[2];
  const float* we0 = (const float*)d_in[3];
  const float* be0 = (const float*)d_in[4];
  const float* we1 = (const float*)d_in[5];
  const float* be1 = (const float*)d_in[6];
  const float* we2 = (const float*)d_in[7];
  const float* be2 = (const float*)d_in[8];
  const float* ge  = (const float*)d_in[9];
  const float* bge = (const float*)d_in[10];
  const float* wn0 = (const float*)d_in[11];
  const float* bn0 = (const float*)d_in[12];
  const float* wn1 = (const float*)d_in[13];
  const float* bn1 = (const float*)d_in[14];
  const float* wn2 = (const float*)d_in[15];
  const float* bn2 = (const float*)d_in[16];
  const float* gn  = (const float*)d_in[17];
  const float* bgn = (const float*)d_in[18];

  float* xnew = (float*)d_out;                       // [N,H]; doubles as agg accumulator
  float* enew = (float*)d_out + (size_t)NNODES * HD; // [E,H]

  unsigned short* ws   = (unsigned short*)d_ws;      // bf16 transposed weights
  unsigned short* we0t = ws;
  unsigned short* we1t = we0t + 384 * 128;
  unsigned short* we2t = we1t + 128 * 128;
  unsigned short* wn0t = we2t + 128 * 128;
  unsigned short* wn1t = wn0t + 256 * 128;
  unsigned short* wn2t = wn1t + 128 * 128;

  // Zero the scatter-add accumulator (= x_new region) each call.
  hipMemsetAsync(xnew, 0, (size_t)NNODES * HD * sizeof(float), stream);

  // Weight convert+transpose to bf16.
  wcvt_kernel<<<(384 * 128 + 255) / 256, 256, 0, stream>>>(we0, we0t, 384);
  wcvt_kernel<<<(128 * 128 + 255) / 256, 256, 0, stream>>>(we1, we1t, 128);
  wcvt_kernel<<<(128 * 128 + 255) / 256, 256, 0, stream>>>(we2, we2t, 128);
  wcvt_kernel<<<(256 * 128 + 255) / 256, 256, 0, stream>>>(wn0, wn0t, 256);
  wcvt_kernel<<<(128 * 128 + 255) / 256, 256, 0, stream>>>(wn1, wn1t, 128);
  wcvt_kernel<<<(128 * 128 + 255) / 256, 256, 0, stream>>>(wn2, wn2t, 128);

  const int edge_smem = (128 * 392 + 128 * 136) * 2;  // 135168 B
  const int node_smem = (128 * 264 + 128 * 136) * 2;  // 102400 B
  (void)hipFuncSetAttribute((const void*)edge_mlp_kernel,
                            hipFuncAttributeMaxDynamicSharedMemorySize, edge_smem);
  (void)hipFuncSetAttribute((const void*)node_mlp_kernel,
                            hipFuncAttributeMaxDynamicSharedMemorySize, node_smem);

  edge_mlp_kernel<<<(NEDGES + 127) / 128, 256, edge_smem, stream>>>(
      x, e, eidx, we0t, we1t, we2t, be0, be1, be2, ge, bge, enew, xnew);

  node_mlp_kernel<<<(NNODES + 127) / 128, 256, node_smem, stream>>>(
      x, wn0t, wn1t, wn2t, bn0, bn1, bn2, gn, bgn, xnew);
}
